// MOE_10728828305811
// MI455X (gfx1250) — matrix operand order repaired and checked by reading, not compiled, not run on hardware
//
#include <hip/hip_runtime.h>
#include <hip/hip_bf16.h>

// MI455X (gfx1250) MoE forward. bf16 WMMA (v_wmma_f32_16x16x32_bf16) for all
// expert GEMMs; fragments pre-packed in workspace; M=32 token tile per WG so
// each B fragment load feeds two WMMAs (halves L2 weight traffic); hidden tile
// in LDS; output accumulated in registers across all 9 experts; x tile staged
// to LDS via the Tensor Data Mover (6-arg builtin form on this toolchain).
// Requires ws_size >= ~95 MB.

typedef __bf16 bf16_t;
typedef __attribute__((ext_vector_type(16))) __bf16 v16bf;
typedef __attribute__((ext_vector_type(8)))  float  v8f;
typedef __attribute__((ext_vector_type(4)))  unsigned int v4u;
typedef __attribute__((ext_vector_type(4)))  int  v4i;
typedef __attribute__((ext_vector_type(8)))  int  v8i;

#if defined(__has_builtin)
#if __has_builtin(__builtin_amdgcn_tensor_load_to_lds) && \
    __has_builtin(__builtin_amdgcn_s_wait_tensorcnt)
#define MOE_USE_TDM 1
#endif
#endif
#ifndef MOE_USE_TDM
#define MOE_USE_TDM 0
#endif

namespace moecfg {
constexpr int E    = 8;
constexpr int D    = 768;
constexpr int H    = 3072;
constexpr int T    = 32 * 196;   // 6272 tokens
constexpr int NEXP = E + 1;      // routed experts + shared expert
constexpr int MT   = T / 16;     // 392 16-row fragment tiles
constexpr int MT2  = T / 32;     // 196 workgroups (M = 32)
constexpr int KD   = D / 32;     // 24  K-tiles for up-proj  (K = D)
constexpr int KH   = H / 32;     // 96  K-tiles for down-proj (K = H)
constexpr int NH   = H / 16;     // 192 N-tiles over hidden
constexpr int ND   = D / 16;     // 48  N-tiles over model dim
}
using namespace moecfg;

// ---------------------------------------------------------------------------
// Fragment index maps (CDNA5 ISA 7.12.2, wave32):
//   A 16x32 bf16:  lane holds row m=lane&15; within-lane half slot s (0..15)
//                  maps to k = (s>>3)*16 + (lane>>4)*8 + (s&7).
//   B 32x16 bf16:  lane holds col n=lane&15; slot s maps to k as for A.
//   C/D 16x16 f32: vgpr v, lane: m = v + 8*(lane>>4), n = lane&15.
// Packed tiles are 512 halves (1024 B): [lane][slot] contiguous per lane.
// ---------------------------------------------------------------------------

__global__ __launch_bounds__(256) void pack_x_kernel(const float* __restrict__ x,
                                                     bf16_t* __restrict__ xA) {
  int idx = blockIdx.x * 256 + threadIdx.x;
  constexpr int N = MT * KD * 512;
  if (idx >= N) return;
  int slot = idx & 15;
  int lane = (idx >> 4) & 31;
  int kt   = (idx >> 9) % KD;
  int mt   = (idx >> 9) / KD;
  int m = lane & 15;
  int k = ((slot >> 3) << 4) | ((lane >> 4) << 3) | (slot & 7);
  xA[idx] = (bf16_t)x[(size_t)(mt * 16 + m) * D + kt * 32 + k];
}

// W1 [E,H,D] (+ Ws1 [H,D] as expert 8) -> B fragments, N over H, K over D.
__global__ __launch_bounds__(256) void pack_b_up(const float* __restrict__ W1,
                                                 const float* __restrict__ Ws1,
                                                 bf16_t* __restrict__ W1B) {
  int idx = blockIdx.x * 256 + threadIdx.x;
  constexpr int N = NEXP * NH * KD * 512;
  if (idx >= N) return;
  int slot = idx & 15;
  int lane = (idx >> 4) & 31;
  int r    = idx >> 9;
  int kt = r % KD;  r /= KD;
  int nt = r % NH;
  int e  = r / NH;
  int k = ((slot >> 3) << 4) | ((lane >> 4) << 3) | (slot & 7);
  int row = nt * 16 + (lane & 15);  // hidden index
  int col = kt * 32 + k;          // d index
  float v = (e < E) ? W1[((size_t)e * H + row) * D + col]
                    : Ws1[(size_t)row * D + col];
  W1B[idx] = (bf16_t)v;
}

// W2 [E,D,H] (+ Ws2 [D,H] as expert 8) -> B fragments, N over D, K over H.
__global__ __launch_bounds__(256) void pack_b_down(const float* __restrict__ W2,
                                                   const float* __restrict__ Ws2,
                                                   bf16_t* __restrict__ W2B) {
  int idx = blockIdx.x * 256 + threadIdx.x;
  constexpr int N = NEXP * ND * KH * 512;
  if (idx >= N) return;
  int slot = idx & 15;
  int lane = (idx >> 4) & 31;
  int r    = idx >> 9;
  int kt = r % KH;  r /= KH;
  int nt = r % ND;
  int e  = r / ND;
  int k = ((slot >> 3) << 4) | ((lane >> 4) << 3) | (slot & 7);
  int row = nt * 16 + (lane & 15);  // d index
  int col = kt * 32 + k;          // hidden index
  float v = (e < E) ? W2[((size_t)e * D + row) * H + col]
                    : Ws2[(size_t)row * H + col];
  W2B[idx] = (bf16_t)v;
}

// ---------------------------------------------------------------------------
// Router: one wave32 per token. logits = x@Wr.T + br + bias; softmax; top-k
// gates kept at their softmax values (not renormalized); shared column = 1.
// ---------------------------------------------------------------------------
__global__ __launch_bounds__(256) void router_kernel(const float* __restrict__ x,
                                                     const float* __restrict__ bias,
                                                     const float* __restrict__ Wr,
                                                     const float* __restrict__ br,
                                                     const int* __restrict__ kptr,
                                                     float* __restrict__ gates) {
  int gwave = (blockIdx.x * blockDim.x + threadIdx.x) >> 5;
  int lane  = threadIdx.x & 31;
  if (gwave >= T) return;
  const float* xt = x + (size_t)gwave * D;
  float acc[E];
#pragma unroll
  for (int e = 0; e < E; ++e) acc[e] = 0.f;
  for (int d = lane; d < D; d += 32) {
    float xv = xt[d];
#pragma unroll
    for (int e = 0; e < E; ++e) acc[e] += xv * Wr[e * D + d];
  }
#pragma unroll
  for (int e = 0; e < E; ++e)
#pragma unroll
    for (int off = 16; off > 0; off >>= 1)
      acc[e] += __shfl_xor(acc[e], off, 32);
  if (lane == 0) {
    float lg[E];
    float mx = -3.402823466e38f;
#pragma unroll
    for (int e = 0; e < E; ++e) {
      lg[e] = acc[e] + br[e] + bias[e];
      mx = fmaxf(mx, lg[e]);
    }
    float s = 0.f;
#pragma unroll
    for (int e = 0; e < E; ++e) { lg[e] = __expf(lg[e] - mx); s += lg[e]; }
    float inv = 1.f / s;
#pragma unroll
    for (int e = 0; e < E; ++e) lg[e] *= inv;
    float g[E];
#pragma unroll
    for (int e = 0; e < E; ++e) g[e] = 0.f;
    int k = *kptr;
    unsigned used = 0;
    for (int j = 0; j < k && j < E; ++j) {
      int bi = -1;
      float bv = -1.f;
#pragma unroll
      for (int e = 0; e < E; ++e)
        if (!((used >> e) & 1u) && lg[e] > bv) { bv = lg[e]; bi = e; }
      used |= 1u << bi;
      g[bi] = bv;
    }
#pragma unroll
    for (int e = 0; e < E; ++e) gates[(size_t)gwave * NEXP + e] = g[e];
    gates[(size_t)gwave * NEXP + E] = 1.0f;
  }
}

// ---------------------------------------------------------------------------
// Main fused kernel: one workgroup (8 waves) per 32-token tile (two 16-row
// A-tiles sharing every B fragment). LDS: x tile (48 KB) + gated-gelu hidden
// tile (192 KB), both in A-fragment order. Output [32 x 768] accumulated in
// registers (12 C-tiles per wave) across all 9 experts.
// ---------------------------------------------------------------------------
__global__ __launch_bounds__(256) void moe_main(const float* __restrict__ gates,
                                                const bf16_t* __restrict__ xA,
                                                const bf16_t* __restrict__ W1B,
                                                const bf16_t* __restrict__ W2B,
                                                const float* __restrict__ b1,
                                                const float* __restrict__ b2,
                                                const float* __restrict__ bs1,
                                                const float* __restrict__ bs2,
                                                float* __restrict__ out) {
  extern __shared__ char smem[];
  bf16_t* lds_x = (bf16_t*)smem;                            // 2*KD*512 halves
  bf16_t* lds_h = (bf16_t*)(smem + (size_t)2 * KD * 1024);  // 2*KH*512 halves

  const int mt2  = blockIdx.x;
  const int tid  = threadIdx.x;
  const int wave = tid >> 5;
  const int lane = tid & 31;
  const int tok0 = mt2 * 32;

  // ---- Stage packed x tile (48 KB contiguous, fragment-ordered) into LDS.
#if MOE_USE_TDM
  if (wave == 0) {
    unsigned long long ga =
        (unsigned long long)(uintptr_t)(xA + (size_t)mt2 * 2 * KD * 512);
    unsigned int lds_off = (unsigned int)(uintptr_t)(void*)lds_x;
    constexpr unsigned NE = (unsigned)(2 * KD * 1024) / 8;  // 6144 x 8B
    v4u g0;
    g0[0] = 1u;                                    // count=1, user mode
    g0[1] = lds_off;                               // lds_addr
    g0[2] = (unsigned int)(ga & 0xffffffffu);      // global_addr[31:0]
    g0[3] = (unsigned int)((ga >> 32) & 0x01ffffffu) | (2u << 30);  // type=2
    v8i g1;
    g1[0] = 3 << 16;                               // data_size = 8 bytes
    g1[1] = (int)((NE & 0xffffu) << 16);           // tensor_dim0 lo16
    g1[2] = (int)((NE >> 16) & 0xffffu) | (1 << 16);  // dim0 hi | tensor_dim1=1
    g1[3] = (int)((NE & 0xffffu) << 16);           // tile_dim0
    g1[4] = 1;                                     // tile_dim1 = 1
    g1[5] = (int)NE;                               // tensor_dim0_stride lo32
    g1[6] = 0;
    g1[7] = 0;
    v4i z4 = {0, 0, 0, 0};
    v8i z8 = {0, 0, 0, 0, 0, 0, 0, 0};
    __builtin_amdgcn_tensor_load_to_lds(g0, g1, z4, z4, z8, 0);
    __builtin_amdgcn_s_wait_tensorcnt(0);
  }
#else
  {
    const uint4* src = (const uint4*)(xA + (size_t)mt2 * 2 * KD * 512);
    uint4* dst = (uint4*)lds_x;
    constexpr int CH = 2 * KD * 1024 / 16;  // 16B chunks
    for (int i = tid; i < CH; i += 256) dst[i] = src[i];
  }
#endif
  __syncthreads();

  v8f acc0[6], acc1[6];
#pragma unroll
  for (int j = 0; j < 6; ++j) { acc0[j] = v8f{}; acc1[j] = v8f{}; }

  for (int e = 0; e < NEXP; ++e) {
    // Gates for this tile/expert (uniform across the block).
    float gv[32];
    float anyg = 0.f;
#pragma unroll
    for (int t = 0; t < 32; ++t) {
      gv[t] = gates[(size_t)(tok0 + t) * NEXP + e];
      anyg += fabsf(gv[t]);
    }
    if (anyg == 0.f) continue;  // uniform skip: expert unused by whole tile

    // ---- Phase 1: hidden = g * gelu(x @ W1e^T + b1e), into LDS (A layout).
    const bf16_t* W1e = W1B + (size_t)e * NH * KD * 512;
    for (int j = 0; j < NH / 8; ++j) {          // 24 n-tiles per wave
      const int nt = wave * (NH / 8) + j;
      v8f cc[2];
      cc[0] = v8f{}; cc[1] = v8f{};
      const bf16_t* Bp = W1e + (size_t)nt * KD * 512 + lane * 16;
      const bf16_t* A0 = lds_x + lane * 16;
      const bf16_t* A1 = lds_x + (size_t)KD * 512 + lane * 16;
      for (int kt = 0; kt < KD; ++kt) {
        v16bf b  = *(const v16bf*)(Bp + (size_t)kt * 512);
        v16bf a0 = *(const v16bf*)(A0 + (size_t)kt * 512);
        v16bf a1 = *(const v16bf*)(A1 + (size_t)kt * 512);
        cc[0] = __builtin_amdgcn_wmma_f32_16x16x32_bf16(false, a0, false, b,
                                                        (short)0, cc[0], false, false);
        cc[1] = __builtin_amdgcn_wmma_f32_16x16x32_bf16(false, a1, false, b,
                                                        (short)0, cc[1], false, false);
      }
      const int n   = lane & 15;
      const int h   = nt * 16 + n;
      const int kt2 = h >> 5;
      const int kk  = h & 31;
      const int lgrp  = ((kk >> 3) & 1) << 4;             // lane group from k
      const int slot2 = ((kk >> 4) << 3) | (kk & 7);
      const float bb = (e < E) ? b1[(size_t)e * H + h] : bs1[h];
#pragma unroll
      for (int mi = 0; mi < 2; ++mi) {
        bf16_t* hb = lds_h + (size_t)mi * KH * 512 + (size_t)kt2 * 512 + slot2;
#pragma unroll
        for (int v = 0; v < 8; ++v) {
          const int m = v + ((lane >> 4) << 3);
          float xv  = cc[mi][v] + bb;
          float gel = 0.5f * xv * (1.f + erff(xv * 0.70710678118654752f));
          float gm  = (lane & 16) ? gv[mi * 16 + v + 8] : gv[mi * 16 + v];
          hb[(size_t)(m + lgrp) * 16] = (bf16_t)(gel * gm);
        }
      }
    }
    __syncthreads();  // hidden tile complete

    // ---- Phase 2: acc += hidden @ W2e^T (gates already folded in).
    const bf16_t* W2e = W2B + (size_t)e * ND * KH * 512;
#pragma unroll
    for (int j = 0; j < 6; ++j) {               // 6 n-tiles (96 d-cols) / wave
      const int nt = wave * 6 + j;
      v8f c0 = acc0[j], c1 = acc1[j];
      const bf16_t* Bp = W2e + (size_t)nt * KH * 512 + lane * 16;
      const bf16_t* A0 = lds_h + lane * 16;
      const bf16_t* A1 = lds_h + (size_t)KH * 512 + lane * 16;
      for (int kt = 0; kt < KH; ++kt) {
        v16bf b  = *(const v16bf*)(Bp + (size_t)kt * 512);
        v16bf a0 = *(const v16bf*)(A0 + (size_t)kt * 512);
        v16bf a1 = *(const v16bf*)(A1 + (size_t)kt * 512);
        c0 = __builtin_amdgcn_wmma_f32_16x16x32_bf16(false, a0, false, b,
                                                     (short)0, c0, false, false);
        c1 = __builtin_amdgcn_wmma_f32_16x16x32_bf16(false, a1, false, b,
                                                     (short)0, c1, false, false);
      }
      acc0[j] = c0; acc1[j] = c1;
    }
    __syncthreads();  // done reading lds_h before next expert overwrites it
  }

  // ---- Epilogue: add sum_e g_e*b2_e + bs2, store fp32 output once.
#pragma unroll
  for (int mi = 0; mi < 2; ++mi) {
    const int tok0m = tok0 + mi * 16;
#pragma unroll
    for (int j = 0; j < 6; ++j) {
      const int dcol = (wave * 6 + j) * 16 + (lane & 15);
      const float bbase = bs2[dcol];
#pragma unroll
      for (int v = 0; v < 8; ++v) {
        const int t = v + ((lane >> 4) << 3);
        float bias = bbase;
#pragma unroll
        for (int e2 = 0; e2 < E; ++e2)
          bias += gates[(size_t)(tok0m + t) * NEXP + e2] * b2[(size_t)e2 * D + dcol];
        float cv = mi ? acc1[j][v] : acc0[j][v];
        out[(size_t)(tok0m + t) * D + dcol] = cv + bias;
      }
    }
  }
}

// ---------------------------------------------------------------------------
extern "C" void kernel_launch(void* const* d_in, const int* in_sizes, int n_in,
                              void* d_out, int out_size, void* d_ws, size_t ws_size,
                              hipStream_t stream) {
  (void)in_sizes; (void)n_in; (void)out_size; (void)ws_size;
  const float* x   = (const float*)d_in[0];
  const float* bia = (const float*)d_in[1];
  const float* Wr  = (const float*)d_in[2];
  const float* br  = (const float*)d_in[3];
  const float* W1  = (const float*)d_in[4];
  const float* b1  = (const float*)d_in[5];
  const float* W2  = (const float*)d_in[6];
  const float* b2  = (const float*)d_in[7];
  const float* Ws1 = (const float*)d_in[8];
  const float* bs1 = (const float*)d_in[9];
  const float* Ws2 = (const float*)d_in[10];
  const float* bs2 = (const float*)d_in[11];
  const int*   kp  = (const int*)d_in[12];
  float* out = (float*)d_out;

  char* ws = (char*)d_ws;
  float*  gates = (float*)ws;
  size_t  off   = (size_t)T * NEXP * sizeof(float);
  bf16_t* xA  = (bf16_t*)(ws + off); off += (size_t)MT * KD * 512 * 2;
  bf16_t* W1B = (bf16_t*)(ws + off); off += (size_t)NEXP * NH * KD * 512 * 2;
  bf16_t* W2B = (bf16_t*)(ws + off);

  {
    constexpr int n = MT * KD * 512;
    pack_x_kernel<<<(n + 255) / 256, 256, 0, stream>>>(x, xA);
  }
  {
    constexpr int n = NEXP * NH * KD * 512;
    pack_b_up<<<(n + 255) / 256, 256, 0, stream>>>(W1, Ws1, W1B);
  }
  {
    constexpr int n = NEXP * ND * KH * 512;
    pack_b_down<<<(n + 255) / 256, 256, 0, stream>>>(W2, Ws2, W2B);
  }
  router_kernel<<<(T * 32 + 255) / 256, 256, 0, stream>>>(x, bia, Wr, br, kp, gates);

  const size_t smem = (size_t)2 * KD * 1024 + (size_t)2 * KH * 1024;  // 245760 B
  moe_main<<<MT2, 256, smem, stream>>>(gates, xA, W1B, W2B, b1, b2, bs1, bs2, out);
}
